// CostVolumePyramid_80762565034124
// MI455X (gfx1250) — matrix operand order repaired and checked by reading, not compiled, not run on hardware
//
#include <hip/hip_runtime.h>

typedef __attribute__((ext_vector_type(16))) __bf16 v16bf;
typedef __attribute__((ext_vector_type(8)))  __bf16 v8bf;
typedef __attribute__((ext_vector_type(8)))  float  v8f;
typedef __attribute__((ext_vector_type(4)))  float  v4f;

__device__ __forceinline__ v16bf cat16(v8bf a, v8bf b) {
    return __builtin_shufflevector(a, b, 0,1,2,3,4,5,6,7,8,9,10,11,12,13,14,15);
}

// 3-term bf16 emulation of fp32 MAC: acc += (ah+al)*(bh+bl), dropping lo*lo
__device__ __forceinline__ v8f wmma3(v16bf ah, v16bf al, v16bf bh, v16bf bl, v8f acc) {
    acc = __builtin_amdgcn_wmma_f32_16x16x32_bf16(false, al, false, bh, (short)0, acc, false, false);
    acc = __builtin_amdgcn_wmma_f32_16x16x32_bf16(false, ah, false, bl, (short)0, acc, false, false);
    acc = __builtin_amdgcn_wmma_f32_16x16x32_bf16(false, ah, false, bh, (short)0, acc, false, false);
    return acc;
}

// Integer-domain truncating bf16 split of a float PAIR, via v_perm_b32.
// ph = {trunc_bf16(f1), trunc_bf16(f0)}, pl = residuals likewise. 6 VALU total.
__device__ __forceinline__ void split2(float f0, float f1, unsigned& ph, unsigned& pl) {
    const unsigned u0 = __builtin_bit_cast(unsigned, f0);
    const unsigned u1 = __builtin_bit_cast(unsigned, f1);
    ph = __builtin_amdgcn_perm(u1, u0, 0x07060302u);       // top halfwords
    const float h0 = __builtin_bit_cast(float, u0 & 0xffff0000u);
    const float h1 = __builtin_bit_cast(float, u1 & 0xffff0000u);
    const unsigned l0 = __builtin_bit_cast(unsigned, f0 - h0);
    const unsigned l1 = __builtin_bit_cast(unsigned, f1 - h1);
    pl = __builtin_amdgcn_perm(l1, l0, 0x07060302u);
}

// single-value split -> packed {hi (low16), lo (high16)}; 4 VALU
__device__ __forceinline__ unsigned pack_split(float v) {
    const unsigned u = __builtin_bit_cast(unsigned, v);
    const float hf = __builtin_bit_cast(float, u & 0xffff0000u);
    const unsigned l = __builtin_bit_cast(unsigned, v - hf);
    return __builtin_amdgcn_perm(l, u, 0x07060302u);       // {l.hi16, u.hi16}
}

// ---------------------------------------------------------------------------
// Correlation cost volume. Block = (b,h,wb); wave delta computes one Gram tile.
// Double-buffered LDS, pipelined. Staging handles even-c channel PAIRS so the
// split is 2x v_perm + 2x v_sub and commits are b32 LDS stores.
// ---------------------------------------------------------------------------
template <int C, int H, int W, int D>
__global__ __launch_bounds__(32 * (D / 16 + 1)) void corr_kernel(
    const float* __restrict__ L, const float* __restrict__ R,
    unsigned* __restrict__ cvp) {
    constexpr int nWb  = W / 16;
    constexpr int nD   = D / 16 + 1;      // delta tiles covering d in [0,D)
    constexpr int SW   = 16 * nD;         // R strip width
    constexpr int NTHR = 32 * nD;
    constexpr int RG   = SW / 4;          // float4 groups per R row
    constexpr int CPR  = NTHR / RG;       // == 8 for all scales (exact)
    constexpr int NCH  = C / 32;
    static_assert(RG * CPR == NTHR, "R staging must be an exact bijection");

    __shared__ __attribute__((aligned(16))) unsigned short sAh[2][16][32];
    __shared__ __attribute__((aligned(16))) unsigned short sAl[2][16][32];
    __shared__ __attribute__((aligned(16))) unsigned short sBh[2][SW][32];
    __shared__ __attribute__((aligned(16))) unsigned short sBl[2][SW][32];

    int t = blockIdx.x;
    const int wb = t % nWb; t /= nWb;
    const int h  = t % H;   t /= H;
    const int b  = t;
    const int w0 = wb * 16;
    const int tid   = threadIdx.x;
    const int lane  = tid & 31;
    const int delta = tid >> 5;
    const int half  = lane >> 4;
    const int ln    = lane & 15;

    const size_t HW = (size_t)H * W;
    const float* Lb = L + (size_t)b * C * HW + (size_t)h * W;
    const float* Rb = R + (size_t)b * C * HW + (size_t)h * W;
    const int wmin = w0 - 16 * (nD - 1);

    // fixed staging roles (divisions hoisted out of all loops)
    // A: threads 0..63 handle (q = w-float4, cp = even channel pair base)
    const int aq  = tid & 3;
    const int acp = tid >> 2;             // 0..15 (valid when tid < 64)
    // R: (rq = strip float4, cp0 + 8r = channel pair), 2 reps cover 16 pairs
    const int rq  = tid % RG;
    const int cp0 = tid / RG;             // 0..CPR-1 (== 0..7)
    const int ws  = wmin + rq * 4;        // strip start; <0 => whole group pads
    const bool rok = (ws >= 0);
    const float* Rbase = Rb + (rok ? ws : 0);   // clamped, always-valid pointer

    v4f la0, la1;                         // A rows 2*acp, 2*acp+1
    v4f rv[2][2];                         // [rep][row of pair]
    auto issue = [&](int k0) {            // global b128 loads -> regs
        if (tid < 64) {
            la0 = *(const v4f*)(Lb + (size_t)(k0 + 2 * acp) * HW + w0 + aq * 4);
            la1 = *(const v4f*)(Lb + (size_t)(k0 + 2 * acp + 1) * HW + w0 + aq * 4);
        }
#pragma unroll
        for (int r = 0; r < 2; ++r) {
            const int c = 2 * (cp0 + r * CPR);
            rv[r][0] = *(const v4f*)(Rbase + (size_t)(k0 + c) * HW);
            rv[r][1] = *(const v4f*)(Rbase + (size_t)(k0 + c + 1) * HW);
        }
    };
    auto commit = [&](int buf) {          // perm-based split + b32 LDS stores
        if (tid < 64) {
#pragma unroll
            for (int e = 0; e < 4; ++e) {
                unsigned ph, pl;
                split2(la0[e], la1[e], ph, pl);
                *(unsigned*)&sAh[buf][aq * 4 + e][2 * acp] = ph;
                *(unsigned*)&sAl[buf][aq * 4 + e][2 * acp] = pl;
            }
        }
#pragma unroll
        for (int r = 0; r < 2; ++r) {
            const int c = 2 * (cp0 + r * CPR);
#pragma unroll
            for (int e = 0; e < 4; ++e) {
                unsigned ph, pl;
                split2(rv[r][0][e], rv[r][1][e], ph, pl);
                if (!rok) { ph = 0u; pl = 0u; }   // zero pad (w < d region)
                *(unsigned*)&sBh[buf][rq * 4 + e][c] = ph;
                *(unsigned*)&sBl[buf][rq * 4 + e][c] = pl;
            }
        }
    };

    issue(0);
    commit(0);
    v8f acc = {};
    const int s = 16 * (nD - 1 - delta) + ln;   // strip index of B column n=ln
#pragma unroll
    for (int cch = 0; cch < NCH; ++cch) {
        if (cch + 1 < NCH) issue((cch + 1) * 32);
        __syncthreads();
        const int buf = cch & 1;
        v16bf ah = cat16(*(const v8bf*)&sAh[buf][ln][half * 16],
                         *(const v8bf*)&sAh[buf][ln][half * 16 + 8]);
        v16bf al = cat16(*(const v8bf*)&sAl[buf][ln][half * 16],
                         *(const v8bf*)&sAl[buf][ln][half * 16 + 8]);
        v16bf bh = cat16(*(const v8bf*)&sBh[buf][s][half * 16],
                         *(const v8bf*)&sBh[buf][s][half * 16 + 8]);
        v16bf bl = cat16(*(const v8bf*)&sBl[buf][s][half * 16],
                         *(const v8bf*)&sBl[buf][s][half * 16 + 8]);
        acc = wmma3(ah, al, bh, bl, acc);
        if (cch + 1 < NCH) commit((cch + 1) & 1);   // other buffer: no race
    }

    const float scale = 1.f / (float)C;   // mean over channels
#pragma unroll
    for (int r = 0; r < 8; ++r) {
        const int m = r + half * 8;
        const int d = 16 * delta + m - ln;
        if (d >= 0 && d < D) {
            const size_t idx = (((size_t)b * D + d) * H + h) * W + (w0 + m);
            cvp[idx] = pack_split(acc[r] * scale);
        }
    }
}

// ---------------------------------------------------------------------------
// Pack conv weights into per-lane WMMA A layout (bf16 hi/lo), once.
// Uses the same truncating split as the cv path.
// ---------------------------------------------------------------------------
template <int CIO>
__global__ __launch_bounds__(64) void pack_weights(const float* __restrict__ Wt,
                                                   unsigned short* __restrict__ ph,
                                                   unsigned short* __restrict__ pl) {
    constexpr int K  = CIO * 9;
    constexpr int NK = (K + 31) / 32;
    constexpr int nOb = CIO / 16;
    const int tid = blockIdx.x * 64 + threadIdx.x;
    if (tid >= nOb * NK * 32) return;
    const int lane = tid & 31;
    const int c    = (tid >> 5) % NK;
    const int ob   = tid / (32 * NK);
    const int ln = lane & 15, half = lane >> 4;
#pragma unroll
    for (int j = 0; j < 16; ++j) {
        const int kA = c * 32 + half * 16 + j;
        const float v = (kA < K) ? Wt[(size_t)(ob * 16 + ln) * K + kA] : 0.f;
        const unsigned p = pack_split(v);
        ph[(size_t)tid * 16 + j] = (unsigned short)(p & 0xffffu);
        pl[(size_t)tid * 16 + j] = (unsigned short)(p >> 16);
    }
}

// ---------------------------------------------------------------------------
// Dilated 3x3 conv as implicit GEMM. Block = (b,y,xb); wave = out-channel blk.
// Pipelined, double-buffered. kl-pair gather merges with 2x v_perm, commits
// as b32 LDS stores; A operands are prepacked (aligned 32B loads).
// ---------------------------------------------------------------------------
template <int CIO, int H, int W, int DIL>
__global__ __launch_bounds__(32 * (CIO / 16)) void conv_kernel(
    const unsigned* __restrict__ cvp,
    const unsigned short* __restrict__ pAh, const unsigned short* __restrict__ pAl,
    float* __restrict__ Y) {
    constexpr int nWb  = W / 16;
    constexpr int nOb  = CIO / 16;
    constexpr int K    = CIO * 9;
    constexpr int NK   = (K + 31) / 32;
    constexpr int NTHR = 32 * nOb;
    constexpr int PPT  = 256 / NTHR;      // kl-pairs per thread
    constexpr int KPS  = NTHR / 16;       // kl-pair stride per rep

    __shared__ __attribute__((aligned(16))) unsigned short sBh[2][16][32];
    __shared__ __attribute__((aligned(16))) unsigned short sBl[2][16][32];

    int t = blockIdx.x;
    const int xb = t % nWb; t /= nWb;
    const int y  = t % H;   t /= H;
    const int b  = t;
    const int x0 = xb * 16;
    const int tid  = threadIdx.x;
    const int lane = tid & 31;
    const int ob   = tid >> 5;
    const int half = lane >> 4;
    const int ln   = lane & 15;
    const int n    = tid & 15;            // staged column: fixed per thread
    const int kp0  = tid >> 4;            // staged kl-pair base: fixed

    const size_t HW   = (size_t)H * W;
    const size_t base = (size_t)b * CIO * HW;

    // one clamped, branch-free im2col gather for k = i*9 + tt
    auto gather = [&](int k, int i, int tt) -> unsigned {
        const int yy = y + DIL * (tt / 3 - 1);
        const int xx = x0 + n + DIL * (tt % 3 - 1);
        const bool ok = (k < K) && ((unsigned)yy < (unsigned)H) &&
                        ((unsigned)xx < (unsigned)W);
        const int ic = (i < CIO) ? i : (CIO - 1);
        const int yc = yy < 0 ? 0 : (yy >= H ? H - 1 : yy);
        const int xc = xx < 0 ? 0 : (xx >= W ? W - 1 : xx);
        const unsigned v = cvp[base + ((size_t)ic * H + yc) * W + xc];
        return ok ? v : 0u;
    };

    unsigned gh[PPT], gl[PPT];
    auto issue = [&](int cch) {
        const int k0 = cch * 32;
#pragma unroll
        for (int r = 0; r < PPT; ++r) {
            const int kl = 2 * (kp0 + r * KPS);
            const int k  = k0 + kl;
            const int i  = k / 9;          // one division per PAIR
            const int tt = k - i * 9;
            const int i2  = (tt == 8) ? i + 1 : i;
            const int tt2 = (tt == 8) ? 0 : tt + 1;
            const unsigned v0 = gather(k, i, tt);
            const unsigned v1 = gather(k + 1, i2, tt2);
            gh[r] = __builtin_amdgcn_perm(v1, v0, 0x05040100u);  // low halves
            gl[r] = __builtin_amdgcn_perm(v1, v0, 0x07060302u);  // high halves
        }
    };
    auto commit = [&](int buf) {
#pragma unroll
        for (int r = 0; r < PPT; ++r) {
            const int kl = 2 * (kp0 + r * KPS);
            *(unsigned*)&sBh[buf][n][kl] = gh[r];
            *(unsigned*)&sBl[buf][n][kl] = gl[r];
        }
    };

    issue(0);
    commit(0);
    v8f acc = {};
#pragma unroll 2
    for (int cch = 0; cch < NK; ++cch) {
        if (cch + 1 < NK) issue(cch + 1);
        __syncthreads();
        const int buf = cch & 1;
        const size_t aoff = ((size_t)(ob * NK + cch) * 32 + lane) * 16;  // 32B-aligned
        v16bf ah = *(const v16bf*)(pAh + aoff);
        v16bf al = *(const v16bf*)(pAl + aoff);
        v16bf bh = cat16(*(const v8bf*)&sBh[buf][ln][half * 16],
                         *(const v8bf*)&sBh[buf][ln][half * 16 + 8]);
        v16bf bl = cat16(*(const v8bf*)&sBl[buf][ln][half * 16],
                         *(const v8bf*)&sBl[buf][ln][half * 16 + 8]);
        acc = wmma3(ah, al, bh, bl, acc);
        if (cch + 1 < NK) commit((cch + 1) & 1);    // other buffer: no race
    }

#pragma unroll
    for (int r = 0; r < 8; ++r) {
        const int m = r + half * 8;
        Y[base + (size_t)(ob * 16 + m) * HW + (size_t)y * W + (x0 + ln)] = acc[r];
    }
}

extern "C" void kernel_launch(void* const* d_in, const int* in_sizes, int n_in,
                              void* d_out, int out_size, void* d_ws, size_t ws_size,
                              hipStream_t stream) {
    (void)in_sizes; (void)n_in; (void)out_size; (void)ws_size;
    const float* left0  = (const float*)d_in[0];
    const float* right0 = (const float*)d_in[1];
    const float* left1  = (const float*)d_in[2];
    const float* right1 = (const float*)d_in[3];
    const float* left2  = (const float*)d_in[4];
    const float* right2 = (const float*)d_in[5];
    const float* W3 = (const float*)d_in[6];
    const float* W2 = (const float*)d_in[7];
    const float* W1 = (const float*)d_in[8];

    float* out = (float*)d_out;
    const size_t n0 = (size_t)4 * 64 * 192 * 320;
    const size_t n1 = (size_t)4 * 32 * 96 * 160;
    const size_t n2 = (size_t)4 * 16 * 48 * 80;
    float* out0 = out;
    float* out1 = out + n0;
    float* out2 = out + n0 + n1;

    // workspace: packed split-bf16 cost volume (L2-resident) + packed weights
    const size_t ncv = n0 + n1 + n2;
    unsigned* cvp  = (unsigned*)d_ws;
    unsigned* cvp0 = cvp;
    unsigned* cvp1 = cvp + n0;
    unsigned* cvp2 = cvp + n0 + n1;
    const size_t p0 = (size_t)4 * 18 * 32 * 16;   // 36864
    const size_t p1 = (size_t)2 * 9  * 32 * 16;   //  9216
    const size_t p2 = (size_t)1 * 5  * 32 * 16;   //  2560
    unsigned short* pAh0 = (unsigned short*)(cvp + ncv);
    unsigned short* pAl0 = pAh0 + p0;
    unsigned short* pAh1 = pAl0 + p0;
    unsigned short* pAl1 = pAh1 + p1;
    unsigned short* pAh2 = pAl1 + p1;
    unsigned short* pAl2 = pAh2 + p2;

    // weight prepack (tiny)
    pack_weights<64><<<(int)((p0 / 16 + 63) / 64), 64, 0, stream>>>(W3, pAh0, pAl0);
    pack_weights<32><<<(int)((p1 / 16 + 63) / 64), 64, 0, stream>>>(W2, pAh1, pAl1);
    pack_weights<16><<<(int)((p2 / 16 + 63) / 64), 64, 0, stream>>>(W1, pAh2, pAl2);

    // correlation cost volumes (banded Gram via pipelined bf16-split WMMA)
    corr_kernel<128, 192, 320, 64><<<4 * 192 * 20, 160, 0, stream>>>(left0, right0, cvp0);
    corr_kernel<128, 96, 160, 32><<<4 * 96 * 10, 96, 0, stream>>>(left1, right1, cvp1);
    corr_kernel<128, 48, 80, 16><<<4 * 48 * 5, 64, 0, stream>>>(left2, right2, cvp2);

    // dilated 3x3 convs (implicit GEMM via WMMA, prepacked A, pipelined B)
    conv_kernel<64, 192, 320, 3><<<4 * 192 * 20, 128, 0, stream>>>(cvp0, pAh0, pAl0, out0);
    conv_kernel<32, 96, 160, 2><<<4 * 96 * 10, 64, 0, stream>>>(cvp1, pAh1, pAl1, out1);
    conv_kernel<16, 48, 80, 1><<<4 * 48 * 5, 32, 0, stream>>>(cvp2, pAh2, pAl2, out2);
}
